// GraphEncoder_15994458210640
// MI455X (gfx1250) — matrix operand order repaired and checked by reading, not compiled, not run on hardware
//
#include <hip/hip_runtime.h>
#include <hip/hip_bf16.h>

// ---------------------------------------------------------------------------
// GraphEncoder for MI455X (gfx1250, wave32, WMMA).
// E=4, H=64, PV=300, RD=64, NS=8, SP=6, T=2, N=5000.
// Dominant work: 16x dense [5000x5000]@[5000x64] -> bf16 WMMA, f32 accum.
// Adjacency is 0/1 -> exact in bf16; one-time cast halves HBM traffic and the
// 200MB bf16 copy nearly fits the 192MB L2 for cross-iteration reuse.
// Async global->LDS staging (ASYNCcnt) for the A^T tiles.
// ---------------------------------------------------------------------------

typedef __attribute__((ext_vector_type(16))) __bf16      v16bf;
typedef __attribute__((ext_vector_type(8)))  float       v8f;
typedef __attribute__((ext_vector_type(4)))  unsigned int u32x4;

union BF16Frag { u32x4 u[2]; v16bf v; };

static __device__ __forceinline__ v8f wmma_bf16(v16bf a, v16bf b, v8f c) {
  // (neg_a, A, neg_b, B, c_mod, C, reuse_a, reuse_b)
  return __builtin_amdgcn_wmma_f32_16x16x32_bf16(false, a, false, b,
                                                 (short)0, c, false, false);
}

// A-fragment element i (of 16 halves) -> K offset within the 32-wide K tile.
// ISA 7.12.2: VGPR0-3 hold K0..15 (lanes0-15 K0-7, lanes16-31 K8-15),
//             VGPR4-7 hold K16..31 (lanes0-15 K16-23, lanes16-31 K24-31).
static __device__ __forceinline__ int a_kmap(int i, int half) {
  return (i & 7) + ((i >> 3) << 4) + (half << 3);
}

// --------------------------- async global->LDS -----------------------------
#if __has_builtin(__builtin_amdgcn_global_load_async_to_lds_b128)
#define HAVE_ASYNC_LDS 1
// Builtin signature (from hipcc diagnostic): (int4 AS1*, int4 AS3*, Ii, Ii)
typedef int vi4 __attribute__((vector_size(16)));
typedef __attribute__((address_space(1))) vi4 vi4_as1;
typedef __attribute__((address_space(3))) vi4 vi4_as3;
template <int OFF>
static __device__ __forceinline__ void async_ldg_b128(const void* g, void* l) {
  __builtin_amdgcn_global_load_async_to_lds_b128((vi4_as1*)g, (vi4_as3*)l,
                                                 OFF, 0);
}
#endif

static __device__ __forceinline__ void wait_async0() {
#if __has_builtin(__builtin_amdgcn_s_wait_asynccnt)
  __builtin_amdgcn_s_wait_asynccnt(0);
#else
  asm volatile("s_wait_asynccnt 0" ::: "memory");
#endif
}

// ---------------------------------------------------------------------------
// h0 = tanh([tanh(nv@W_red+b_red) | node_states | size_pos] @ W_init + b_init)
// one block (64 threads) per node
// ---------------------------------------------------------------------------
__global__ __launch_bounds__(64) void init_kernel(
    const float* __restrict__ nv, const float* __restrict__ ns,
    const float* __restrict__ sp, const float* __restrict__ W_red,
    const float* __restrict__ b_red, const float* __restrict__ W_init,
    const float* __restrict__ b_init, float* __restrict__ h_f32,
    __bf16* __restrict__ h_bf, int N) {
  __shared__ float sh[80];
  const int n = blockIdx.x;
  const int j = threadIdx.x;
  float acc = b_red[j];
  const float* nvr = nv + (size_t)n * 300;
#pragma unroll 4
  for (int k = 0; k < 300; ++k) acc += nvr[k] * W_red[k * 64 + j];
  sh[j] = tanhf(acc);
  if (j < 8) sh[64 + j] = ns[(size_t)n * 8 + j];
  if (j < 6) sh[72 + j] = sp[(size_t)n * 6 + j];
  __syncthreads();
  float a2 = b_init[j];
#pragma unroll
  for (int k = 0; k < 78; ++k) a2 += sh[k] * W_init[k * 64 + j];
  float hv = tanhf(a2);
  h_f32[(size_t)n * 64 + j] = hv;
  h_bf[(size_t)n * 64 + j] = (__bf16)hv;
}

// ---------------------------------------------------------------------------
// one-time adjacency fp32 -> bf16 (exact: A is 0/1)
// ---------------------------------------------------------------------------
__global__ void a2b_kernel(const float* __restrict__ A,
                           __bf16* __restrict__ Ab, size_t total) {
  size_t i = (size_t)blockIdx.x * blockDim.x + threadIdx.x;
  const size_t stride = (size_t)gridDim.x * blockDim.x;
  for (; i < total; i += stride) Ab[i] = (__bf16)A[i];
}

// ---------------------------------------------------------------------------
// hs[d][e] = tanh(h @ W[d][e] + b[d][e])  -> bf16 [2][4][N][64]
// one wave per 16-row x 64-col tile, K=64 in two WMMA k-steps
// ---------------------------------------------------------------------------
__global__ __launch_bounds__(32) void hs_kernel(
    const float* __restrict__ h, const float* __restrict__ W_out,
    const float* __restrict__ b_out, const float* __restrict__ W_in,
    const float* __restrict__ b_in, __bf16* __restrict__ hs, int N) {
  const int n0 = blockIdx.x * 16, e = blockIdx.y, d = blockIdx.z;
  const float* W = (d == 0 ? W_out : W_in) + (size_t)e * 64 * 64;
  const float* bias = (d == 0 ? b_out : b_in) + e * 64;
  const int l = threadIdx.x, half = l >> 4, lm = l & 15;
  const v8f vzero = {0.f, 0.f, 0.f, 0.f, 0.f, 0.f, 0.f, 0.f};
  v8f acc[4] = {vzero, vzero, vzero, vzero};
  const int row = (n0 + lm < N) ? (n0 + lm) : (N - 1);
  const bool rok = (n0 + lm) < N;
  for (int k0 = 0; k0 < 64; k0 += 32) {
    BF16Frag a;
    const float* hr = h + (size_t)row * 64 + k0;
#pragma unroll
    for (int i = 0; i < 16; ++i)
      a.v[i] = (__bf16)(rok ? hr[a_kmap(i, half)] : 0.0f);
#pragma unroll
    for (int jt = 0; jt < 4; ++jt) {
      BF16Frag b;
      const float* wr = W + (size_t)k0 * 64 + jt * 16 + lm;
#pragma unroll
      for (int i = 0; i < 16; ++i) b.v[i] = (__bf16)wr[(size_t)a_kmap(i, half) * 64];
      acc[jt] = wmma_bf16(a.v, b.v, acc[jt]);
    }
  }
  __bf16* out = hs + (size_t)(d * 4 + e) * N * 64;
  const int rbase = n0 + (half ? 8 : 0);
#pragma unroll
  for (int jt = 0; jt < 4; ++jt)
#pragma unroll
    for (int r = 0; r < 8; ++r) {
      const int rr = rbase + r;
      if (rr < N) {
        const float v = acc[jt][r] + bias[jt * 16 + lm];
        out[(size_t)rr * 64 + jt * 16 + lm] = (__bf16)tanhf(v);
      }
    }
}

// ---------------------------------------------------------------------------
// msg[n][d*256+e*64+j] = (d==0 ? A[e] : A[e]^T) @ hs[d][e]
// one wave per 16x64 tile; dir1 stages the A^T tile through LDS.
// bf16-A + async path: each lane async-copies its 32B row slice to LDS,
// s_wait_asynccnt, then reads the transposed fragment as bf16 (no cvt).
// OOB columns/rows only contaminate fragment rows >= N (dropped at the store)
// or multiply zeroed B fragments -> harmless; Ab has a 256B tail pad.
// ---------------------------------------------------------------------------
template <bool ABF16>
__global__ __launch_bounds__(32) void msg_kernel(
    const void* __restrict__ Aptr, const __bf16* __restrict__ hs,
    __bf16* __restrict__ msg, int N) {
  __shared__ float tile[32][17];                   // fp32 fallback staging
#if defined(HAVE_ASYNC_LDS)
  __shared__ alignas(16) __bf16 tileb[32][16];     // bf16 async staging
#endif
  const int n0 = blockIdx.x * 16, e = blockIdx.y, d = blockIdx.z;
  const int l = threadIdx.x, half = l >> 4, lm = l & 15;
  const size_t NN = (size_t)N * N;
  const float*  Af = (const float*)Aptr + (size_t)e * NN;
  const __bf16* Ab = (const __bf16*)Aptr + (size_t)e * NN;
  const __bf16* hsb = hs + (size_t)(d * 4 + e) * N * 64;
  const v8f vzero = {0.f, 0.f, 0.f, 0.f, 0.f, 0.f, 0.f, 0.f};
  v8f acc[4] = {vzero, vzero, vzero, vzero};
  const int rowA = (n0 + lm < N) ? (n0 + lm) : (N - 1);
  const bool colfull = (n0 + 16) <= N;
  const int kmain = (N / 32) * 32;

  for (int k0 = 0; k0 < kmain; k0 += 32) {
    BF16Frag a;
    if (d == 0) {
      if (ABF16) {
        const __bf16* p = Ab + (size_t)rowA * N + k0 + half * 8;
        const u32x4* q = (const u32x4*)p;  // 16B aligned (N*2 % 16 == 0)
        a.u[0] = q[0];        // K = k0+half*8 .. +7
        a.u[1] = q[2];        // K = k0+16+half*8 .. +7
        if (k0 + 64 <= kmain) __builtin_prefetch(p + 32, 0, 1);
      } else {
        const float* p = Af + (size_t)rowA * N + k0;
#pragma unroll
        for (int i = 0; i < 16; ++i) a.v[i] = (__bf16)p[a_kmap(i, half)];
      }
    } else {
      const int krow = k0 + l;  // < k0+32 <= kmain <= N in main loop
#if defined(HAVE_ASYNC_LDS)
      if (ABF16) {
        const __bf16* src = Ab + (size_t)krow * N + n0;  // 32B per lane
        async_ldg_b128<0>(src, &tileb[l][0]);
        async_ldg_b128<16>(src, &tileb[l][0]);
        wait_async0();
        __syncthreads();
#pragma unroll
        for (int i = 0; i < 16; ++i) a.v[i] = tileb[a_kmap(i, half)][lm];
        __syncthreads();
      } else
#endif
      {
        if (ABF16) {
          const __bf16* p = Ab + (size_t)krow * N + n0;
          if (colfull) {
#pragma unroll
            for (int i = 0; i < 16; ++i) tile[l][i] = (float)p[i];
          } else {
#pragma unroll
            for (int i = 0; i < 16; ++i)
              tile[l][i] = (n0 + i < N) ? (float)p[i] : 0.0f;
          }
        } else {
          const float* p = Af + (size_t)krow * N + n0;
          if (colfull) {
#pragma unroll
            for (int i = 0; i < 16; ++i) tile[l][i] = p[i];
          } else {
#pragma unroll
            for (int i = 0; i < 16; ++i)
              tile[l][i] = (n0 + i < N) ? p[i] : 0.0f;
          }
        }
        __syncthreads();
#pragma unroll
        for (int i = 0; i < 16; ++i)
          a.v[i] = (__bf16)tile[a_kmap(i, half)][lm];
        __syncthreads();
      }
    }
    const __bf16* hq = hsb + (size_t)k0 * 64 + lm;
#pragma unroll
    for (int jt = 0; jt < 4; ++jt) {
      BF16Frag b;
#pragma unroll
      for (int i = 0; i < 16; ++i)
        b.v[i] = hq[(size_t)a_kmap(i, half) * 64 + jt * 16];
      acc[jt] = wmma_bf16(a.v, b.v, acc[jt]);
    }
  }

  // K tail (N % 32 == 8): rows >= N feed zeroed B fragments, so garbage in
  // the matching A-fragment K positions is multiplied by zero.
  if (kmain < N) {
    const int k0 = kmain;
    BF16Frag a;
    if (d == 0) {
#pragma unroll
      for (int i = 0; i < 16; ++i) {
        const int k = k0 + a_kmap(i, half);
        float v = 0.0f;
        if (k < N)
          v = ABF16 ? (float)Ab[(size_t)rowA * N + k] : Af[(size_t)rowA * N + k];
        a.v[i] = (__bf16)v;
      }
    } else {
      const int krow = k0 + l;
#if defined(HAVE_ASYNC_LDS)
      if (ABF16) {
        if (krow < N) {  // EXEC-masked async issue (legal; only WMMA needs full EXEC)
          const __bf16* src = Ab + (size_t)krow * N + n0;
          async_ldg_b128<0>(src, &tileb[l][0]);
          async_ldg_b128<16>(src, &tileb[l][0]);
        }
        wait_async0();
        __syncthreads();
#pragma unroll
        for (int i = 0; i < 16; ++i) a.v[i] = tileb[a_kmap(i, half)][lm];
      } else
#endif
      {
#pragma unroll
        for (int i = 0; i < 16; ++i) {
          float v = 0.0f;
          if (krow < N && (n0 + i) < N)
            v = ABF16 ? (float)Ab[(size_t)krow * N + n0 + i]
                      : Af[(size_t)krow * N + n0 + i];
          tile[l][i] = v;
        }
        __syncthreads();
#pragma unroll
        for (int i = 0; i < 16; ++i)
          a.v[i] = (__bf16)tile[a_kmap(i, half)][lm];
      }
    }
#pragma unroll
    for (int jt = 0; jt < 4; ++jt) {
      BF16Frag b;
#pragma unroll
      for (int i = 0; i < 16; ++i) {
        const int kb = k0 + a_kmap(i, half);
        b.v[i] = (kb < N) ? hsb[(size_t)kb * 64 + jt * 16 + lm] : (__bf16)0.0f;
      }
      acc[jt] = wmma_bf16(a.v, b.v, acc[jt]);
    }
  }

  const int rbase = n0 + (half ? 8 : 0);
#pragma unroll
  for (int jt = 0; jt < 4; ++jt)
#pragma unroll
    for (int r = 0; r < 8; ++r) {
      const int row = rbase + r;
      if (row < N)
        msg[(size_t)row * 512 + d * 256 + e * 64 + jt * 16 + lm] =
            (__bf16)acc[jt][r];
    }
}

// ---------------------------------------------------------------------------
// G[N,384] = [ msg@W_ih_rz | h@W_hh_rz | msg@W_ih_c | h@W_hh_c ]
// ---------------------------------------------------------------------------
__global__ __launch_bounds__(32) void gru_gemm_kernel(
    const __bf16* __restrict__ msg, const __bf16* __restrict__ hb,
    const float* __restrict__ W_ih_rz, const float* __restrict__ W_hh_rz,
    const float* __restrict__ W_ih_c, const float* __restrict__ W_hh_c,
    float* __restrict__ G, int N) {
  const int n0 = blockIdx.x * 16;
  const int jt = blockIdx.y;  // 0..23 (384/16 column tiles)
  const __bf16* X; const float* W; int Kdim, ldW, colbase;
  if (jt < 8)       { X = msg; Kdim = 512; W = W_ih_rz; ldW = 128; colbase = jt * 16; }
  else if (jt < 16) { X = hb;  Kdim = 64;  W = W_hh_rz; ldW = 128; colbase = (jt - 8) * 16; }
  else if (jt < 20) { X = msg; Kdim = 512; W = W_ih_c;  ldW = 64;  colbase = (jt - 16) * 16; }
  else              { X = hb;  Kdim = 64;  W = W_hh_c;  ldW = 64;  colbase = (jt - 20) * 16; }
  const int xld = (Kdim == 512) ? 512 : 64;
  const int l = threadIdx.x, half = l >> 4, lm = l & 15;
  const int row = (n0 + lm < N) ? (n0 + lm) : (N - 1);
  v8f acc = {0.f, 0.f, 0.f, 0.f, 0.f, 0.f, 0.f, 0.f};
  for (int k0 = 0; k0 < Kdim; k0 += 32) {
    BF16Frag a;
    const u32x4* q = (const u32x4*)(X + (size_t)row * xld + k0 + half * 8);
    a.u[0] = q[0];
    a.u[1] = q[2];
    BF16Frag b;
    const float* wr = W + (size_t)k0 * ldW + colbase + lm;
#pragma unroll
    for (int i = 0; i < 16; ++i) b.v[i] = (__bf16)wr[(size_t)a_kmap(i, half) * ldW];
    acc = wmma_bf16(a.v, b.v, acc);
  }
  const int rbase = n0 + (half ? 8 : 0);
#pragma unroll
  for (int r = 0; r < 8; ++r) {
    const int rr = rbase + r;
    if (rr < N) G[(size_t)rr * 384 + jt * 16 + lm] = acc[r];
  }
}

// ---------------------------------------------------------------------------
// LayerNorms + GRU gating; one 128-thread block per node
// ---------------------------------------------------------------------------
static __device__ __forceinline__ float block_sum(float v, float* red, int t) {
  red[t] = v;
  __syncthreads();
  for (int s = 64; s > 0; s >>= 1) {
    if (t < s) red[t] += red[t + s];
    __syncthreads();
  }
  const float r = red[0];
  __syncthreads();
  return r;
}

__global__ __launch_bounds__(128) void gru_pointwise(
    const float* __restrict__ G,
    const float* b_ih_rz, const float* b_hh_rz,
    const float* b_ih_c, const float* b_hh_c,
    const float* g_i2h, const float* be_i2h,
    const float* g_h2h, const float* be_h2h,
    const float* g_ci, const float* be_ci,
    const float* g_ch, const float* be_ch,
    float* __restrict__ h, __bf16* __restrict__ h_bf, int N) {
  __shared__ float red[128];
  __shared__ float shy[128];
  const int n = blockIdx.x, t = threadIdx.x;
  const float* Gr = G + (size_t)n * 384;
  const float inv128 = 1.0f / 128.0f, inv64 = 1.0f / 64.0f, eps = 1e-5f;

  const float v1 = Gr[t] + b_ih_rz[t];
  const float m1 = block_sum(v1, red, t) * inv128;
  const float q1 = block_sum(v1 * v1, red, t) * inv128;
  const float ln1 = (v1 - m1) * rsqrtf(q1 - m1 * m1 + eps) * g_i2h[t] + be_i2h[t];

  const float v2 = Gr[128 + t] + b_hh_rz[t];
  const float m2 = block_sum(v2, red, t) * inv128;
  const float q2 = block_sum(v2 * v2, red, t) * inv128;
  const float ln2 = (v2 - m2) * rsqrtf(q2 - m2 * m2 + eps) * g_h2h[t] + be_h2h[t];

  shy[t] = 1.0f / (1.0f + expf(-(ln1 + ln2)));
  __syncthreads();

  const float v3 = (t < 64) ? (Gr[256 + t] + b_ih_c[t]) : 0.0f;
  const float m3 = block_sum(v3, red, t) * inv64;
  const float q3 = block_sum(v3 * v3, red, t) * inv64;
  const float v4 = (t < 64) ? (Gr[320 + t] + b_hh_c[t]) : 0.0f;
  const float m4 = block_sum(v4, red, t) * inv64;
  const float q4 = block_sum(v4 * v4, red, t) * inv64;

  if (t < 64) {
    const float ln3 = (v3 - m3) * rsqrtf(q3 - m3 * m3 + eps) * g_ci[t] + be_ci[t];
    const float ln4 = (v4 - m4) * rsqrtf(q4 - m4 * m4 + eps) * g_ch[t] + be_ch[t];
    const float r = shy[t], z = shy[64 + t];
    const float c = tanhf(ln3 + r * ln4);
    const float h0 = h[(size_t)n * 64 + t];
    const float hn = z * h0 + (1.0f - z) * c;
    h[(size_t)n * 64 + t] = hn;
    h_bf[(size_t)n * 64 + t] = (__bf16)hn;
  }
}

__global__ void copy_kernel(const float* __restrict__ src,
                            float* __restrict__ dst, int n) {
  const int i = blockIdx.x * blockDim.x + threadIdx.x;
  if (i < n) dst[i] = src[i];
}

// ---------------------------------------------------------------------------
extern "C" void kernel_launch(void* const* d_in, const int* in_sizes, int n_in,
                              void* d_out, int out_size, void* d_ws,
                              size_t ws_size, hipStream_t stream) {
  const float* A       = (const float*)d_in[0];
  const float* nstates = (const float*)d_in[1];
  // d_in[2] = node_ids (unused)
  const float* nvec    = (const float*)d_in[3];
  const float* nsp     = (const float*)d_in[4];
  const float* W_red   = (const float*)d_in[5];
  const float* b_red   = (const float*)d_in[6];
  const float* W_init  = (const float*)d_in[7];
  const float* b_init  = (const float*)d_in[8];
  const float* W_out   = (const float*)d_in[9];
  const float* b_out   = (const float*)d_in[10];
  const float* W_in_e  = (const float*)d_in[11];
  const float* b_in_e  = (const float*)d_in[12];
  const float* W_ih_rz = (const float*)d_in[13];
  const float* b_ih_rz = (const float*)d_in[14];
  const float* W_hh_rz = (const float*)d_in[15];
  const float* b_hh_rz = (const float*)d_in[16];
  const float* W_ih_c  = (const float*)d_in[17];
  const float* b_ih_c  = (const float*)d_in[18];
  const float* W_hh_c  = (const float*)d_in[19];
  const float* b_hh_c  = (const float*)d_in[20];
  const float* g_i2h   = (const float*)d_in[21];
  const float* be_i2h  = (const float*)d_in[22];
  const float* g_h2h   = (const float*)d_in[23];
  const float* be_h2h  = (const float*)d_in[24];
  const float* g_ci    = (const float*)d_in[25];
  const float* be_ci   = (const float*)d_in[26];
  const float* g_ch    = (const float*)d_in[27];
  const float* be_ch   = (const float*)d_in[28];
  const int N = 5000;

  char* ws = (char*)d_ws;
  size_t off = 0;
  auto take = [&](size_t bytes) -> void* {
    void* p = ws + off;
    off = (off + bytes + 255) & ~(size_t)255;
    return p;
  };
  float*  h_f32 = (float*) take((size_t)N * 64 * 4);
  __bf16* h_bf  = (__bf16*)take((size_t)N * 64 * 2);
  __bf16* hs    = (__bf16*)take((size_t)8 * N * 64 * 2);
  __bf16* msg   = (__bf16*)take((size_t)N * 512 * 2);
  float*  G     = (float*) take((size_t)N * 384 * 4);
  const size_t a_elems = (size_t)4 * N * N;
  __bf16* Ab = nullptr;
  if (off + a_elems * 2 + 256 <= ws_size)
    Ab = (__bf16*)take(a_elems * 2 + 256);  // 256B tail pad for edge overreads

  init_kernel<<<N, 64, 0, stream>>>(nvec, nstates, nsp, W_red, b_red, W_init,
                                    b_init, h_f32, h_bf, N);
  if (Ab) a2b_kernel<<<4096, 256, 0, stream>>>(A, Ab, a_elems);

  const dim3 mgrid((N + 15) / 16, 4, 2);
  const dim3 ggrid((N + 15) / 16, 24);
  for (int t = 0; t < 2; ++t) {
    hs_kernel<<<mgrid, 32, 0, stream>>>(h_f32, W_out, b_out, W_in_e, b_in_e,
                                        hs, N);
    if (Ab)
      msg_kernel<true><<<mgrid, 32, 0, stream>>>((const void*)Ab, hs, msg, N);
    else
      msg_kernel<false><<<mgrid, 32, 0, stream>>>((const void*)A, hs, msg, N);
    gru_gemm_kernel<<<ggrid, 32, 0, stream>>>(msg, h_bf, W_ih_rz, W_hh_rz,
                                              W_ih_c, W_hh_c, G, N);
    gru_pointwise<<<N, 128, 0, stream>>>(G, b_ih_rz, b_hh_rz, b_ih_c, b_hh_c,
                                         g_i2h, be_i2h, g_h2h, be_h2h, g_ci,
                                         be_ci, g_ch, be_ch, h_f32, h_bf, N);
  }
  copy_kernel<<<(N * 64 + 255) / 256, 256, 0, stream>>>(h_f32, (float*)d_out,
                                                        N * 64);
  (void)in_sizes; (void)n_in; (void)out_size;
}
